// PE_GCL_86930138071709
// MI455X (gfx1250) — matrix operand order repaired and checked by reading, not compiled, not run on hardware
//
#include <hip/hip_runtime.h>

#define N_NODES   50000
#define N_EDGES   800000
#define NF        128

typedef __attribute__((ext_vector_type(16))) __bf16 v16bf;
typedef __attribute__((ext_vector_type(8)))  __bf16 v8bf;
typedef __attribute__((ext_vector_type(8)))  float  v8f;
typedef __attribute__((ext_vector_type(4)))  float  v4f;

// Hardware f32 -> bf16 convert (fptrunc lowers to v_cvt_*_bf16_f32 on gfx1250,
// and the backend can pack pairs with v_cvt_pk_bf16_f32).
__device__ __forceinline__ __bf16 f2bf(float f) { return (__bf16)f; }

__device__ __forceinline__ float siluf(float v) { return v / (1.0f + __expf(-v)); }

// Convert a 128x128 (K x N, row-major) f32 weight into WMMA B-fragment order (bf16).
// Fragment addressing: frag(kc,nt) is 32 lanes x 16 contiguous bf16.
// Lane holds col n = 16*nt + (lane&15); element i holds K = 32*kc + 8*(lane>>4) + (i&7) + 16*(i>>3).
__device__ __forceinline__ void stage_weight_bfrag(__bf16* dst, const float* __restrict__ src, int tid) {
  for (int idx = tid; idx < NF * NF; idx += 256) {
    const int k = idx >> 7, n = idx & 127;
    const int kc = k >> 5, kk = k & 31;
    const int half = (kk >> 3) & 1, i = (kk & 7) | ((kk >> 4) << 3);
    const int lane = half * 16 + (n & 15);
    const int nt = n >> 4;
    dst[(((kc * 8 + nt) * 32 + lane) << 4) + i] = f2bf(src[(k << 7) + n]);
  }
}

__device__ __forceinline__ v16bf load_bfrag(const __bf16* mat, int kc, int nt, int lane) {
  const v8bf* p = (const v8bf*)(mat + (((kc * 8 + nt) * 32 + lane) << 4));
  v8bf lo = p[0];
  v8bf hv = p[1];
  return __builtin_shufflevector(lo, hv, 0,1,2,3,4,5,6,7,8,9,10,11,12,13,14,15);
}

// A-fragment (16x32 bf16) from a global f32 row: row m = lane&15 supplied by caller,
// K = 32*kc + (lane&16 ? 8:0) + {0..7, 16..23}  -> two contiguous 8-float segments.
__device__ __forceinline__ v16bf load_afrag_g(const float* __restrict__ rowp, int kc, int lane) {
  const float* p = rowp + kc * 32 + ((lane & 16) ? 8 : 0);
  v4f f0 = *(const v4f*)(p + 0);
  v4f f1 = *(const v4f*)(p + 4);
  v4f f2 = *(const v4f*)(p + 16);
  v4f f3 = *(const v4f*)(p + 20);
  v16bf a;
#pragma unroll
  for (int j = 0; j < 4; ++j) {
    a[j]      = f2bf(f0[j]);
    a[4 + j]  = f2bf(f1[j]);
    a[8 + j]  = f2bf(f2[j]);
    a[12 + j] = f2bf(f3[j]);
  }
  return a;
}

// Same A-fragment pattern but from an LDS bf16 row (activation staging tile).
__device__ __forceinline__ v16bf load_afrag_l(const __bf16* rowp, int kc, int lane) {
  const __bf16* p = rowp + kc * 32 + ((lane & 16) ? 8 : 0);
  v8bf lo = *(const v8bf*)(p);
  v8bf hv = *(const v8bf*)(p + 16);
  return __builtin_shufflevector(lo, hv, 0,1,2,3,4,5,6,7,8,9,10,11,12,13,14,15);
}

#define WMMA_BF16(acc, a, b) \
  (acc) = __builtin_amdgcn_wmma_f32_16x16x32_bf16(false, (a), false, (b), (short)0, (acc), false, false)

// One K-chunk of a 16x(8*16) GEMM row: 8 WMMAs against one A fragment, with the
// B-fragment LDS loads software-pipelined one N-tile ahead so the scheduler can
// issue partial dscnt waits instead of a full drain per WMMA.
__device__ __forceinline__ void wmma_row(v8f* acc, v16bf a, const __bf16* mat, int kc, int lane) {
  v16bf b0 = load_bfrag(mat, kc, 0, lane);
#pragma unroll
  for (int nt = 0; nt < 7; ++nt) {
    v16bf b1 = load_bfrag(mat, kc, nt + 1, lane);
    WMMA_BF16(acc[nt], a, b0);
    b0 = b1;
  }
  WMMA_BF16(acc[7], a, b0);
}

// ---------------------------------------------------------------------------
// Edge kernel: per wave, one 16-edge tile. Fused:
//   geometry -> edge MLP (h_row@W1a + h_col@W1b + radial*w1r + be1, silu, @We2+be2, silu)
//   -> atomic scatter into msg, and coord gate (@Wc1+bc1, silu, @Wc2) -> coord atomics.
// LDS: [0,131072) bf16 B-frags (W1a,W1b,We2,Wc1), [131072,133632) f32 vecs
//      (be1,be2,bc1,w1r,wc2), [133632,166400) per-wave 16x128 bf16 staging,
//      [166400,169472) per-wave edge metadata.
// ---------------------------------------------------------------------------
__global__ __launch_bounds__(256, 1)
void egcl_edge_kernel(const float* __restrict__ h, const float* __restrict__ x,
                      const long long* __restrict__ ei, const float* __restrict__ box,
                      const float* __restrict__ We1, const float* __restrict__ be1,
                      const float* __restrict__ We2, const float* __restrict__ be2,
                      const float* __restrict__ Wc1, const float* __restrict__ bc1,
                      const float* __restrict__ Wc2,
                      float* __restrict__ msg, float* __restrict__ cupd)
{
  extern __shared__ char smem[];
  __bf16* wf   = (__bf16*)smem;
  float*  vec  = (float*)(smem + 131072);
  __bf16* stgA = (__bf16*)(smem + 133632);
  char*   meta = smem + 166400;

  const int tid  = threadIdx.x;
  const int wv   = tid >> 5;
  const int lane = tid & 31;
  const int sub  = lane & 15;
  const int hi   = lane >> 4;

  stage_weight_bfrag(wf + 0 * 16384, We1,           tid);   // rows 0..127   (h[row] half)
  stage_weight_bfrag(wf + 1 * 16384, We1 + NF * NF, tid);   // rows 128..255 (h[col] half)
  stage_weight_bfrag(wf + 2 * 16384, We2,           tid);
  stage_weight_bfrag(wf + 3 * 16384, Wc1,           tid);
  if (tid < 128) {
    vec[tid]       = be1[tid];
    vec[128 + tid] = be2[tid];
    vec[256 + tid] = bc1[tid];
    vec[384 + tid] = We1[256 * NF + tid];   // radial row of We1
    vec[512 + tid] = Wc2[tid];
  }
  __syncthreads();

  __bf16* stage = stgA + wv * (16 * NF);
  int*    mrow  = (int*)(meta + wv * 384);
  int*    mcol  = mrow + 16;
  float*  mcdx  = (float*)(mcol + 16);
  float*  mcdy  = mcdx + 16;
  float*  mcdz  = mcdy + 16;
  float*  mrad  = mcdz + 16;

  const int e0 = (blockIdx.x * 8 + wv) * 16;

  if (lane < 16) {
    const int e  = e0 + lane;
    const int r_ = (int)ei[e];
    const int c_ = (int)ei[N_EDGES + e];
    const float bx = box[0], by = box[1], bz = box[2];
    float dx = x[r_ * 3 + 0] - x[c_ * 3 + 0];
    float dy = x[r_ * 3 + 1] - x[c_ * 3 + 1];
    float dz = x[r_ * 3 + 2] - x[c_ * 3 + 2];
    dx -= bx * rintf(dx / bx);                 // jnp.round == round-half-even
    dy -= by * rintf(dy / by);
    dz -= bz * rintf(dz / bz);
    const float r2  = dx * dx + dy * dy + dz * dz;
    const float inv = 1.0f / (sqrtf(r2 + 1e-8f) + 1.0f);
    mrow[lane] = r_;  mcol[lane] = c_;
    mcdx[lane] = dx * inv;  mcdy[lane] = dy * inv;  mcdz[lane] = dz * inv;
    mrad[lane] = sqrtf(r2);
  }
  __syncthreads();

  float rad8[8];
#pragma unroll
  for (int r = 0; r < 8; ++r) rad8[r] = mrad[r + 8 * hi];

  // ---- layer 1: acc = be1 + radial*w1r + h_row@W1a + h_col@W1b ----
  v8f acc[8];
#pragma unroll
  for (int nt = 0; nt < 8; ++nt) {
    const float b  = vec[nt * 16 + sub];
    const float wr = vec[384 + nt * 16 + sub];
#pragma unroll
    for (int r = 0; r < 8; ++r) acc[nt][r] = __builtin_fmaf(rad8[r], wr, b);
  }

  const int ri = mrow[sub], ci = mcol[sub];
  const float* hr = h + (size_t)ri * NF;
  const float* hc = h + (size_t)ci * NF;

#pragma unroll
  for (int kc = 0; kc < 4; ++kc) {
    v16bf a0 = load_afrag_g(hr, kc, lane);
    wmma_row(acc, a0, wf + 0 * 16384, kc, lane);
    v16bf a1 = load_afrag_g(hc, kc, lane);
    wmma_row(acc, a1, wf + 1 * 16384, kc, lane);
  }

  // silu + stage as 16x128 bf16 (C layout -> A layout transpose via LDS)
#pragma unroll
  for (int nt = 0; nt < 8; ++nt)
#pragma unroll
    for (int r = 0; r < 8; ++r)
      stage[(r + 8 * hi) * NF + nt * 16 + sub] = f2bf(siluf(acc[nt][r]));

  // ---- layer 2: edge_feat = silu(act1 @ We2 + be2) ----
#pragma unroll
  for (int nt = 0; nt < 8; ++nt) {
    const float b = vec[128 + nt * 16 + sub];
#pragma unroll
    for (int r = 0; r < 8; ++r) acc[nt][r] = b;
  }
  const __bf16* arow = stage + sub * NF;
#pragma unroll
  for (int kc = 0; kc < 4; ++kc) {
    v16bf a = load_afrag_l(arow, kc, lane);
    wmma_row(acc, a, wf + 2 * 16384, kc, lane);
  }

  // silu -> scatter into msg (fire-and-forget f32 atomics) + restage for coord gate
  int rows8[8];
#pragma unroll
  for (int r = 0; r < 8; ++r) rows8[r] = mrow[r + 8 * hi];

#pragma unroll
  for (int nt = 0; nt < 8; ++nt) {
#pragma unroll
    for (int r = 0; r < 8; ++r) {
      const float v = siluf(acc[nt][r]);
      stage[(r + 8 * hi) * NF + nt * 16 + sub] = f2bf(v);
      atomicAdd(msg + (size_t)rows8[r] * NF + nt * 16 + sub, v);
    }
  }

  // ---- coord gate: scale = silu(edge_feat @ Wc1 + bc1) @ Wc2 ----
  v8f accC[8];
#pragma unroll
  for (int nt = 0; nt < 8; ++nt) {
    const float b = vec[256 + nt * 16 + sub];
#pragma unroll
    for (int r = 0; r < 8; ++r) accC[nt][r] = b;
  }
#pragma unroll
  for (int kc = 0; kc < 4; ++kc) {
    v16bf a = load_afrag_l(arow, kc, lane);
    wmma_row(accC, a, wf + 3 * 16384, kc, lane);
  }
  float part[8];
#pragma unroll
  for (int r = 0; r < 8; ++r) part[r] = 0.0f;
#pragma unroll
  for (int nt = 0; nt < 8; ++nt) {
    const float w = vec[512 + nt * 16 + sub];
#pragma unroll
    for (int r = 0; r < 8; ++r) part[r] += siluf(accC[nt][r]) * w;
  }
  // reduce the N dimension across the 16 lanes of each half-wave
#pragma unroll
  for (int r = 0; r < 8; ++r) {
    float p = part[r];
    p += __shfl_xor(p, 1, 16);
    p += __shfl_xor(p, 2, 16);
    p += __shfl_xor(p, 4, 16);
    p += __shfl_xor(p, 8, 16);
    part[r] = p;
  }
  if (sub < 8) {
    const int m = 8 * hi + sub;        // lanes 0..7 -> edges 0..7, lanes 16..23 -> edges 8..15
    float s = part[0];
#pragma unroll
    for (int r = 1; r < 8; ++r) s = (sub == r) ? part[r] : s;
    const int rw = mrow[m];
    atomicAdd(cupd + rw * 3 + 0, mcdx[m] * s);
    atomicAdd(cupd + rw * 3 + 1, mcdy[m] * s);
    atomicAdd(cupd + rw * 3 + 2, mcdz[m] * s);
  }
}

// ---------------------------------------------------------------------------
// Node kernel: h_out = silu([h,msg]@Wn1 + bn1)@Wn2 + bn2 ; x_out = x + coord_update.
// ---------------------------------------------------------------------------
__global__ __launch_bounds__(256, 1)
void egcl_node_kernel(const float* __restrict__ h, const float* __restrict__ x,
                      const float* __restrict__ Wn1, const float* __restrict__ bn1,
                      const float* __restrict__ Wn2, const float* __restrict__ bn2,
                      const float* __restrict__ msg, const float* __restrict__ cupd,
                      float* __restrict__ hout, float* __restrict__ xout)
{
  extern __shared__ char smem[];
  __bf16* wf   = (__bf16*)smem;
  float*  vec  = (float*)(smem + 98304);
  __bf16* stgA = (__bf16*)(smem + 99328);

  const int tid = threadIdx.x, wv = tid >> 5, lane = tid & 31;
  const int sub = lane & 15, hi = lane >> 4;

  stage_weight_bfrag(wf,         Wn1,           tid);   // rows 0..127   (h half)
  stage_weight_bfrag(wf + 16384, Wn1 + NF * NF, tid);   // rows 128..255 (msg half)
  stage_weight_bfrag(wf + 32768, Wn2,           tid);
  if (tid < 128) { vec[tid] = bn1[tid]; vec[128 + tid] = bn2[tid]; }
  __syncthreads();

  const int tile = blockIdx.x * 8 + wv;
  if (tile >= N_NODES / 16) return;
  const int n0 = tile * 16;
  __bf16* stage = stgA + wv * (16 * NF);

  const float* hr = h   + (size_t)(n0 + sub) * NF;
  const float* mr = msg + (size_t)(n0 + sub) * NF;

  v8f acc[8];
#pragma unroll
  for (int nt = 0; nt < 8; ++nt) {
    const float b = vec[nt * 16 + sub];
#pragma unroll
    for (int r = 0; r < 8; ++r) acc[nt][r] = b;
  }
#pragma unroll
  for (int kc = 0; kc < 4; ++kc) {
    v16bf a0 = load_afrag_g(hr, kc, lane);
    wmma_row(acc, a0, wf, kc, lane);
    v16bf a1 = load_afrag_g(mr, kc, lane);
    wmma_row(acc, a1, wf + 16384, kc, lane);
  }

#pragma unroll
  for (int nt = 0; nt < 8; ++nt)
#pragma unroll
    for (int r = 0; r < 8; ++r)
      stage[(r + 8 * hi) * NF + nt * 16 + sub] = f2bf(siluf(acc[nt][r]));

#pragma unroll
  for (int nt = 0; nt < 8; ++nt) {
    const float b = vec[128 + nt * 16 + sub];
#pragma unroll
    for (int r = 0; r < 8; ++r) acc[nt][r] = b;
  }
  const __bf16* arow = stage + sub * NF;
#pragma unroll
  for (int kc = 0; kc < 4; ++kc) {
    v16bf a = load_afrag_l(arow, kc, lane);
    wmma_row(acc, a, wf + 32768, kc, lane);
  }

#pragma unroll
  for (int nt = 0; nt < 8; ++nt)
#pragma unroll
    for (int r = 0; r < 8; ++r)
      hout[(size_t)(n0 + r + 8 * hi) * NF + nt * 16 + sub] = acc[nt][r];

  if (lane < 16) {
    const int nd = n0 + lane;
    xout[nd * 3 + 0] = x[nd * 3 + 0] + cupd[nd * 3 + 0];
    xout[nd * 3 + 1] = x[nd * 3 + 1] + cupd[nd * 3 + 1];
    xout[nd * 3 + 2] = x[nd * 3 + 2] + cupd[nd * 3 + 2];
  }
}

__global__ void zero_f32(float* __restrict__ p, int n) {
  const int i = blockIdx.x * blockDim.x + threadIdx.x;
  if (i < n) p[i] = 0.0f;
}

extern "C" void kernel_launch(void* const* d_in, const int* in_sizes, int n_in,
                              void* d_out, int out_size, void* d_ws, size_t ws_size,
                              hipStream_t stream) {
  (void)in_sizes; (void)n_in; (void)out_size; (void)ws_size;

  const float*     h    = (const float*)d_in[0];
  const float*     x    = (const float*)d_in[1];
  const long long* ei   = (const long long*)d_in[2];   // reference dtype is int64
  const float*     box  = (const float*)d_in[3];
  const float*     We1  = (const float*)d_in[4];
  const float*     be1  = (const float*)d_in[5];
  const float*     We2  = (const float*)d_in[6];
  const float*     be2  = (const float*)d_in[7];
  const float*     Wn1  = (const float*)d_in[8];
  const float*     bn1  = (const float*)d_in[9];
  const float*     Wn2  = (const float*)d_in[10];
  const float*     bn2  = (const float*)d_in[11];
  const float*     Wc1  = (const float*)d_in[12];
  const float*     bc1  = (const float*)d_in[13];
  const float*     Wc2  = (const float*)d_in[14];

  float* msg  = (float*)d_ws;                        // [N_NODES, NF]
  float* cupd = msg + (size_t)N_NODES * NF;          // [N_NODES, 3]
  float* hout = (float*)d_out;                       // [N_NODES, NF]
  float* xout = hout + (size_t)N_NODES * NF;         // [N_NODES, 3]

  const size_t lds_edge = 131072 + 2560 + 32768 + 3072;   // 169472 B
  const size_t lds_node = 98304 + 1024 + 32768;           // 132096 B
  hipFuncSetAttribute((const void*)egcl_edge_kernel,
                      hipFuncAttributeMaxDynamicSharedMemorySize, (int)lds_edge);
  hipFuncSetAttribute((const void*)egcl_node_kernel,
                      hipFuncAttributeMaxDynamicSharedMemorySize, (int)lds_node);

  const int nzero = N_NODES * NF + N_NODES * 3;
  zero_f32<<<(nzero + 255) / 256, 256, 0, stream>>>(msg, nzero);

  egcl_edge_kernel<<<N_EDGES / 128, 256, lds_edge, stream>>>(
      h, x, ei, box, We1, be1, We2, be2, Wc1, bc1, Wc2, msg, cupd);

  const int ntiles = N_NODES / 16;                    // 3125
  egcl_node_kernel<<<(ntiles + 7) / 8, 256, lds_node, stream>>>(
      h, x, Wn1, bn1, Wn2, bn2, msg, cupd, hout, xout);
}
